// SAModule_Transformer_4037269258363
// MI455X (gfx1250) — matrix operand order repaired and checked by reading, not compiled, not run on hardware
//
#include <hip/hip_runtime.h>
#include <hip/hip_bf16.h>

// ---------------------------------------------------------------------------
// SA-module + PointTransformerConv for MI455X (gfx1250, wave32, WMMA).
//
// Pipeline:
//   0) swizzle kernels : one-time fp32->bf16 conversion AND permutation of x
//                        and all weights into exact WMMA operand order, so the
//                        hot kernels fetch each operand as one aligned 32-byte
//                        vector load instead of 16 scalar u16 loads.
//   1) fps_kernel      : farthest point sampling, 1 block / cloud
//   2) proj_kernel     : x@{w_lin,w_src,w_dst} via v_wmma_f32_16x16x32_bf16
//   3) delta0_kernel   : pos-MLP at zero offset (self-loop delta)
//   4) knn_kernel      : radius-limited top-K=64 per sampled point
//   5) selfloop_kernel : out[d>=4096] = xw_lin[d] + delta0 (1-edge softmax == 1)
//   6) attn_kernel     : per-dst fused edge MLP + attn projection (both WMMA),
//                        LDS staged in operand-swizzled order, masked softmax
//                        + aggregation.
// ---------------------------------------------------------------------------

#define B_CLOUDS 8
#define P_PTS    2048
#define DIMP     6
#define N_PTS    (B_CLOUDS * P_PTS)     // 16384
#define S_PER    512
#define NSAMP    (B_CLOUDS * S_PER)     // 4096
#define KNB      64
#define C_IN     64
#define C_OUT    128
#define H_POS    64
#define R2       0.25f
#define E_ROWS   80                     // 65 edges padded to 5 x 16 WMMA M-tiles
#define E_VALID  65

typedef __attribute__((ext_vector_type(16))) __bf16 v16bf;
typedef __attribute__((ext_vector_type(8)))  float  v8f;

__device__ __forceinline__ __bf16 f2bf(float f) {
  unsigned u = __builtin_bit_cast(unsigned, f);
  u = u + 0x7FFFu + ((u >> 16) & 1u);               // round-to-nearest-even
  unsigned short s = (unsigned short)(u >> 16);
  return __builtin_bit_cast(__bf16, s);
}
__device__ __forceinline__ float bf2f(__bf16 h) {
  unsigned short s = __builtin_bit_cast(unsigned short, h);
  unsigned u = ((unsigned)s) << 16;
  return __builtin_bit_cast(float, u);
}

// A-operand swizzle (16-bit 16x32 tile, ISA 7.12.2): element (row, k) of an
// M-strip maps to chunk [(strip*nks + ks)*32 + lane][elem] with:
//   lane = ((k>>3)&1)*16 + (row&15)
//   elem = 2*(((k&31)>>4)*4 + ((k&7)>>1)) + (k&1)
__device__ __forceinline__ int a_swz_elem(int k) {
  int kk = k & 31;
  int p  = ((kk >> 4) << 2) + ((kk & 7) >> 1);
  return 2 * p + (kk & 1);
}
__device__ __forceinline__ int a_swz_lane(int row, int k) {
  return (((k >> 3) & 1) << 4) + (row & 15);
}

// ---------------------------------------------------------------------------
// 0a) B-operand swizzle: W[Kdim,128] fp32 -> bf16 chunks
//     dst[((nt*(Kdim/32) + ks)*32 + hlf*16 + l16)*16 + e]
//     with k = ks*32 + (e>>3)*16 + hlf*8 + (e&7), n = nt*16 + l16.
// ---------------------------------------------------------------------------
__global__ __launch_bounds__(256) void swz_b_kernel(const float* __restrict__ W,
                                                    __bf16* __restrict__ dst,
                                                    int Kdim) {
  int idx = blockIdx.x * 256 + threadIdx.x;
  if (idx >= Kdim * C_OUT) return;
  int k = idx / C_OUT, n = idx % C_OUT;
  int nt = n >> 4, l16 = n & 15;
  int ks = k >> 5, hlf = (k >> 3) & 1, e = (((k >> 4) & 1) << 3) + (k & 7);
  int nks = Kdim >> 5;
  dst[(((nt * nks + ks) * 32) + hlf * 16 + l16) * 16 + e] = f2bf(W[idx]);
}

// ---------------------------------------------------------------------------
// 0b) A-operand swizzle for x [16384,64] fp32 -> bf16 chunks per 16-row strip.
// ---------------------------------------------------------------------------
__global__ __launch_bounds__(256) void swz_a_kernel(const float* __restrict__ x,
                                                    __bf16* __restrict__ dst) {
  int idx = blockIdx.x * 256 + threadIdx.x;
  if (idx >= N_PTS * C_IN) return;
  int row = idx / C_IN, k = idx % C_IN;
  int strip = row >> 4, ks = k >> 5;
  dst[(((strip * 2 + ks) * 32) + a_swz_lane(row, k)) * 16 + a_swz_elem(k)] = f2bf(x[idx]);
}

// ---------------------------------------------------------------------------
// 1) Farthest point sampling: one block per cloud, sequential 512-step argmax.
// ---------------------------------------------------------------------------
__global__ __launch_bounds__(256) void fps_kernel(const float* __restrict__ pos,
                                                  int* __restrict__ idx_sel) {
  const int b = blockIdx.x;
  const int tid = threadIdx.x;
  __shared__ float d[P_PTS];
  __shared__ float rv[256];
  __shared__ int   ri[256];
  __shared__ int   s_last;
  for (int i = tid; i < P_PTS; i += 256) d[i] = __builtin_inff();
  if (tid == 0) { s_last = 0; idx_sel[b * S_PER] = 0; }
  __syncthreads();
  for (int it = 1; it < S_PER; ++it) {
    const int last = s_last;
    float lp[DIMP];
#pragma unroll
    for (int j = 0; j < DIMP; ++j) lp[j] = pos[(b * P_PTS + last) * DIMP + j];
    float bv = -1.0f; int bi = 0;
    for (int i = tid; i < P_PTS; i += 256) {
      float dist = 0.f;
#pragma unroll
      for (int j = 0; j < DIMP; ++j) {
        float t = pos[(b * P_PTS + i) * DIMP + j] - lp[j];
        dist += t * t;
      }
      float nd = fminf(d[i], dist);
      d[i] = nd;
      if (nd > bv) { bv = nd; bi = i; }
    }
    rv[tid] = bv; ri[tid] = bi;
    __syncthreads();
    for (int s = 128; s > 0; s >>= 1) {
      if (tid < s) {
        float ov = rv[tid + s]; int oi = ri[tid + s];
        if (ov > rv[tid] || (ov == rv[tid] && oi < ri[tid])) { rv[tid] = ov; ri[tid] = oi; }
      }
      __syncthreads();
    }
    if (tid == 0) { s_last = ri[0]; idx_sel[b * S_PER + it] = ri[0]; }
    __syncthreads();
  }
}

// ---------------------------------------------------------------------------
// 2) Projections: [16384,64] x [64,128] x3 with bf16 WMMA.
//    All operands pre-swizzled: one v16bf vector load each.
// ---------------------------------------------------------------------------
__global__ __launch_bounds__(256) void proj_kernel(
    const v16bf* __restrict__ x_a,
    const v16bf* __restrict__ wl_b, const v16bf* __restrict__ wsrc_b,
    const v16bf* __restrict__ wd_b,
    float* __restrict__ xw_lin, float* __restrict__ xw_src,
    float* __restrict__ xw_dst) {
  const int strip = blockIdx.x;
  const int row0  = strip * 16;
  const int tid   = threadIdx.x;
  const int wave  = tid >> 5, lane = tid & 31;
  const int hlf   = lane >> 4, l16 = lane & 15;

  v16bf a0 = x_a[(strip * 2 + 0) * 32 + lane];
  v16bf a1 = x_a[(strip * 2 + 1) * 32 + lane];

  for (int t = wave; t < 24; t += 8) {
    const int mat = t >> 3, nt = t & 7;
    const v16bf* W = (mat == 0) ? wl_b : (mat == 1) ? wsrc_b : wd_b;
    float*       O = (mat == 0) ? xw_lin : (mat == 1) ? xw_src : xw_dst;
    v8f acc = {};
    acc = __builtin_amdgcn_wmma_f32_16x16x32_bf16(false, a0, false,
              W[(nt * 2 + 0) * 32 + lane], (short)0, acc, false, false);
    acc = __builtin_amdgcn_wmma_f32_16x16x32_bf16(false, a1, false,
              W[(nt * 2 + 1) * 32 + lane], (short)0, acc, false, false);
#pragma unroll
    for (int v = 0; v < 8; ++v)       // C/D: M = v + 8*half, N = lane%16
      O[(row0 + hlf * 8 + v) * C_OUT + nt * 16 + l16] = acc[v];
  }
}

// ---------------------------------------------------------------------------
// 3) delta0 = relu(relu(b1) @ w2 + b2)  (pos-MLP at zero relative position)
// ---------------------------------------------------------------------------
__global__ void delta0_kernel(const float* __restrict__ b1,
                              const float* __restrict__ w2,
                              const float* __restrict__ b2,
                              float* __restrict__ delta0) {
  int c = threadIdx.x;
  if (c < C_OUT) {
    float acc = b2[c];
    for (int h = 0; h < H_POS; ++h) acc += fmaxf(b1[h], 0.f) * w2[h * C_OUT + c];
    delta0[c] = fmaxf(acc, 0.f);
  }
}

// ---------------------------------------------------------------------------
// 4) Radius-limited top-K: one block per sampled point, iterative argmin.
// ---------------------------------------------------------------------------
__global__ __launch_bounds__(256) void knn_kernel(const float* __restrict__ pos,
                                                  const int* __restrict__ idx_sel,
                                                  int* __restrict__ nbr) {
  const int q = blockIdx.x;           // 0..4095 == dst row index
  const int b = q / S_PER;
  const int tid = threadIdx.x;
  __shared__ float d2[P_PTS];
  __shared__ float rv[256];
  __shared__ int   ri[256];
  const int qi = idx_sel[q];
  float qp[DIMP];
#pragma unroll
  for (int j = 0; j < DIMP; ++j) qp[j] = pos[(b * P_PTS + qi) * DIMP + j];
  for (int i = tid; i < P_PTS; i += 256) {
    float dist = 0.f;
#pragma unroll
    for (int j = 0; j < DIMP; ++j) {
      float t = pos[(b * P_PTS + i) * DIMP + j] - qp[j];
      dist += t * t;
    }
    d2[i] = (dist <= R2) ? dist : __builtin_inff();
  }
  __syncthreads();
  for (int k = 0; k < KNB; ++k) {
    float bv = __builtin_inff(); int bi = P_PTS;
    for (int i = tid; i < P_PTS; i += 256) {
      float v = d2[i];
      if (v < bv) { bv = v; bi = i; }
    }
    rv[tid] = bv; ri[tid] = bi;
    __syncthreads();
    for (int s = 128; s > 0; s >>= 1) {
      if (tid < s) {
        float ov = rv[tid + s]; int oi = ri[tid + s];
        if (ov < rv[tid] || (ov == rv[tid] && oi < ri[tid])) { rv[tid] = ov; ri[tid] = oi; }
      }
      __syncthreads();
    }
    if (tid == 0) {
      if (rv[0] < __builtin_inff()) {
        nbr[q * KNB + k] = b * P_PTS + ri[0];
        d2[ri[0]] = __builtin_inff();
      } else {
        nbr[q * KNB + k] = -1;
      }
    }
    __syncthreads();
  }
}

// ---------------------------------------------------------------------------
// 5) Nodes >= 4096 only have their self-loop: single-edge softmax == 1.
// ---------------------------------------------------------------------------
__global__ __launch_bounds__(256) void selfloop_kernel(
    const float* __restrict__ xw_lin, const float* __restrict__ delta0,
    float* __restrict__ out) {
  long long idx = (long long)blockIdx.x * 256 + threadIdx.x;
  const long long total = (long long)(N_PTS - NSAMP) * C_OUT;
  if (idx < total) {
    int node = NSAMP + (int)(idx / C_OUT);
    int c = (int)(idx % C_OUT);
    out[(long long)node * C_OUT + c] = xw_lin[(long long)node * C_OUT + c] + delta0[c];
  }
}

// ---------------------------------------------------------------------------
// 6) Fused per-destination attention. LDS regions are time-multiplexed
//    (WMMA results staged in registers across barriers):
//      [0,2560)      s_rel   f32 [80][8]          (stages A-B)
//      [2560,12800)  s_h     bf16, A-op swizzled  (stages B-C)
//      [0,20480)     s_delta bf16 [80][128]       (stage C-store onward)
//      [20480,40960) s_apre  bf16, A-op swizzled  (stages D-E)
//      [20480,61440) s_a     f32  [80][128]       (stage E-store onward)
// ---------------------------------------------------------------------------
__global__ __launch_bounds__(256) void attn_kernel(
    const float* __restrict__ pos, const int* __restrict__ nbr,
    const float* __restrict__ xw_lin, const float* __restrict__ xw_src,
    const float* __restrict__ xw_dst,
    const float* __restrict__ w1, const float* __restrict__ b1,
    const v16bf* __restrict__ w2_b, const float* __restrict__ b2,
    const v16bf* __restrict__ wattn_b, const float* __restrict__ b_attn,
    float* __restrict__ out) {
  const int d = blockIdx.x;           // destination row, 0..4095
  const int tid = threadIdx.x;
  const int wave = tid >> 5, lane = tid & 31;
  const int hlf = lane >> 4, l16 = lane & 15;

  __shared__ char smem[61440];
  float*  s_rel   = (float*)smem;
  __bf16* s_h     = (__bf16*)(smem + 2560);       // A-op swizzled [5][2][32][16]
  const v16bf* s_hv = (const v16bf*)(smem + 2560);
  __bf16* s_delta = (__bf16*)smem;                // row-major [80][128]
  __bf16* s_apre  = (__bf16*)(smem + 20480);      // A-op swizzled [5][4][32][16]
  const v16bf* s_aprev = (const v16bf*)(smem + 20480);
  float*  s_a     = (float*)(smem + 20480);       // row-major [80][128]
  __shared__ int s_src[E_ROWS];
  __shared__ int s_valid[E_ROWS];

  // --- Stage A: edge list + relative positions ----------------------------
  if (tid < E_ROWS) {
    int e = tid, src = d, valid = 0;
    if (e < KNB) {
      int nv = nbr[d * KNB + e];
      valid = (nv >= 0) && (nv != d);   // remove_self_loops on raw indices
      src = valid ? nv : d;
    } else if (e == KNB) {              // explicit self loop (always valid)
      src = d; valid = 1;
    }
    s_src[e] = src; s_valid[e] = valid;
#pragma unroll
    for (int j = 0; j < DIMP; ++j)
      s_rel[e * 8 + j] = pos[d * DIMP + j] - pos[src * DIMP + j];
    s_rel[e * 8 + 6] = 0.f; s_rel[e * 8 + 7] = 0.f;
    __builtin_prefetch((const void*)(xw_src + (long long)src * C_OUT), 0, 1);
    __builtin_prefetch((const void*)(xw_lin + (long long)src * C_OUT), 0, 1);
  }
  __syncthreads();

  // --- Stage B: h = relu(rel @ w1 + b1); store A-op swizzled ---------------
  for (int idx = tid; idx < E_ROWS * H_POS; idx += 256) {
    int e = idx >> 6, hc = idx & 63;
    float acc = b1[hc];
#pragma unroll
    for (int j = 0; j < DIMP; ++j) acc += s_rel[e * 8 + j] * w1[j * H_POS + hc];
    int mt = e >> 4, ks = hc >> 5;
    s_h[(((mt * 2 + ks) * 32) + a_swz_lane(e, hc)) * 16 + a_swz_elem(hc)] =
        f2bf(fmaxf(acc, 0.f));
  }
  __syncthreads();

  // --- Stage C: delta = relu(h @ w2 + b2)  [80,128], 40 WMMA tiles ---------
  {
    v8f acc[5];
    const v8f z = {0.f, 0.f, 0.f, 0.f, 0.f, 0.f, 0.f, 0.f};
#pragma unroll
    for (int i = 0; i < 5; ++i) acc[i] = z;
#pragma unroll
    for (int i = 0; i < 5; ++i) {
      const int ti = wave * 5 + i;
      const int mt = ti >> 3, nt = ti & 7;
#pragma unroll
      for (int ks = 0; ks < 2; ++ks)
        acc[i] = __builtin_amdgcn_wmma_f32_16x16x32_bf16(
            false, s_hv[(mt * 2 + ks) * 32 + lane],
            false, w2_b[(nt * 2 + ks) * 32 + lane],
            (short)0, acc[i], false, false);
    }
    __syncthreads();                    // all s_h reads done before overwrite
#pragma unroll
    for (int i = 0; i < 5; ++i) {
      const int ti = wave * 5 + i;
      const int mt = ti >> 3, nt = ti & 7;
#pragma unroll
      for (int v = 0; v < 8; ++v) {
        const int r = mt * 16 + hlf * 8 + v, c = nt * 16 + l16;
        s_delta[r * C_OUT + c] = f2bf(fmaxf(acc[i][v] + b2[c], 0.f));
      }
    }
  }
  __syncthreads();

  // --- Stage D: a_pre = xw_dst[d] - xw_src[src] + delta; A-op swizzled -----
  for (int idx = tid; idx < E_ROWS * C_OUT; idx += 256) {
    int e = idx >> 7, c = idx & 127;
    float v = xw_dst[d * C_OUT + c] - xw_src[(long long)s_src[e] * C_OUT + c]
              + bf2f(s_delta[idx]);
    int mt = e >> 4, ks = c >> 5;
    s_apre[(((mt * 4 + ks) * 32) + a_swz_lane(e, c)) * 16 + a_swz_elem(c)] = f2bf(v);
  }
  __syncthreads();

  // --- Stage E: a = a_pre @ w_attn + b_attn  [80,128], K=128 ---------------
  {
    v8f acc[5];
    const v8f z = {0.f, 0.f, 0.f, 0.f, 0.f, 0.f, 0.f, 0.f};
#pragma unroll
    for (int i = 0; i < 5; ++i) acc[i] = z;
#pragma unroll
    for (int i = 0; i < 5; ++i) {
      const int ti = wave * 5 + i;
      const int mt = ti >> 3, nt = ti & 7;
#pragma unroll
      for (int ks = 0; ks < 4; ++ks)
        acc[i] = __builtin_amdgcn_wmma_f32_16x16x32_bf16(
            false, s_aprev[(mt * 4 + ks) * 32 + lane],
            false, wattn_b[(nt * 4 + ks) * 32 + lane],
            (short)0, acc[i], false, false);
    }
    __syncthreads();                    // all s_apre reads done before overwrite
#pragma unroll
    for (int i = 0; i < 5; ++i) {
      const int ti = wave * 5 + i;
      const int mt = ti >> 3, nt = ti & 7;
#pragma unroll
      for (int v = 0; v < 8; ++v) {
        const int r = mt * 16 + hlf * 8 + v, c = nt * 16 + l16;
        s_a[r * C_OUT + c] = acc[i][v] + b_attn[c];
      }
    }
  }
  __syncthreads();

  // --- Stage F: masked softmax over edges + weighted aggregation -----------
  if (tid < C_OUT) {
    const int c = tid;
    float m = -3.0e38f;
    for (int e = 0; e < E_VALID; ++e)
      if (s_valid[e]) m = fmaxf(m, s_a[e * C_OUT + c]);
    float sum = 0.f;
    for (int e = 0; e < E_VALID; ++e) {
      float v = s_valid[e] ? __expf(s_a[e * C_OUT + c] - m) : 0.f;
      s_a[e * C_OUT + c] = v;
      sum += v;
    }
    float acc = 0.f;
    for (int e = 0; e < E_VALID; ++e) {
      if (s_valid[e]) {
        float attn = s_a[e * C_OUT + c] / sum;
        acc += attn * (xw_lin[(long long)s_src[e] * C_OUT + c] + bf2f(s_delta[e * C_OUT + c]));
      }
    }
    out[(long long)d * C_OUT + c] = acc;
  }
}

// ---------------------------------------------------------------------------
extern "C" void kernel_launch(void* const* d_in, const int* in_sizes, int n_in,
                              void* d_out, int out_size, void* d_ws, size_t ws_size,
                              hipStream_t stream) {
  const float* x      = (const float*)d_in[0];
  const float* pos    = (const float*)d_in[1];
  /* d_in[2] = batch: layout is known (b = i / P), unused */
  const float* w_lin  = (const float*)d_in[3];
  const float* w_src  = (const float*)d_in[4];
  const float* w_dst  = (const float*)d_in[5];
  const float* w1     = (const float*)d_in[6];
  const float* b1     = (const float*)d_in[7];
  const float* w2     = (const float*)d_in[8];
  const float* b2     = (const float*)d_in[9];
  const float* w_attn = (const float*)d_in[10];
  const float* b_attn = (const float*)d_in[11];
  float* out = (float*)d_out;

  // Workspace layout (~28.4 MB), all bf16 regions 32B-aligned.
  char*  ws      = (char*)d_ws;
  int*   idx_sel = (int*)ws;                               //  4096 ints
  int*   nbr     = (int*)(ws + 16384);                     //  4096*64 ints
  float* xw_lin  = (float*)(ws + 16384 + 1048576);         //  16384*128 f32
  float* xw_src  = xw_lin + (size_t)N_PTS * C_OUT;
  float* xw_dst  = xw_src + (size_t)N_PTS * C_OUT;
  float* delta0  = xw_dst + (size_t)N_PTS * C_OUT;         //  128 f32
  char*  bfbase  = (char*)(delta0 + 128);
  __bf16* x_a      = (__bf16*)bfbase;                      //  16384*64 (A-swz)
  __bf16* wl_b     = x_a + (size_t)N_PTS * C_IN;           //  64*128   (B-swz)
  __bf16* wsrc_b   = wl_b + C_IN * C_OUT;
  __bf16* wd_b     = wsrc_b + C_IN * C_OUT;
  __bf16* w2_b     = wd_b + C_IN * C_OUT;                  //  64*128   (B-swz)
  __bf16* wattn_b  = w2_b + H_POS * C_OUT;                 //  128*128  (B-swz)

  // 0) one-time bf16 conversion + operand swizzle
  swz_a_kernel<<<(N_PTS * C_IN + 255) / 256, 256, 0, stream>>>(x, x_a);
  swz_b_kernel<<<(C_IN * C_OUT + 255) / 256, 256, 0, stream>>>(w_lin, wl_b, C_IN);
  swz_b_kernel<<<(C_IN * C_OUT + 255) / 256, 256, 0, stream>>>(w_src, wsrc_b, C_IN);
  swz_b_kernel<<<(C_IN * C_OUT + 255) / 256, 256, 0, stream>>>(w_dst, wd_b, C_IN);
  swz_b_kernel<<<(H_POS * C_OUT + 255) / 256, 256, 0, stream>>>(w2, w2_b, H_POS);
  swz_b_kernel<<<(C_OUT * C_OUT + 255) / 256, 256, 0, stream>>>(w_attn, wattn_b, C_OUT);

  fps_kernel<<<B_CLOUDS, 256, 0, stream>>>(pos, idx_sel);
  proj_kernel<<<N_PTS / 16, 256, 0, stream>>>((const v16bf*)x_a, (const v16bf*)wl_b,
                                              (const v16bf*)wsrc_b, (const v16bf*)wd_b,
                                              xw_lin, xw_src, xw_dst);
  delta0_kernel<<<1, 128, 0, stream>>>(b1, w2, b2, delta0);
  knn_kernel<<<NSAMP, 256, 0, stream>>>(pos, idx_sel, nbr);
  {
    long long total = (long long)(N_PTS - NSAMP) * C_OUT;
    int blocks = (int)((total + 255) / 256);
    selfloop_kernel<<<blocks, 256, 0, stream>>>(xw_lin, delta0, out);
  }
  attn_kernel<<<NSAMP, 256, 0, stream>>>(pos, nbr, xw_lin, xw_src, xw_dst,
                                         w1, b1, (const v16bf*)w2_b, b2,
                                         (const v16bf*)wattn_b, b_attn, out);
}
